// SmallSMBlock3DPY_44753559224604
// MI455X (gfx1250) — matrix operand order repaired and checked by reading, not compiled, not run on hardware
//
#include <hip/hip_runtime.h>
#include <stdint.h>

typedef _Float16 v16h  __attribute__((ext_vector_type(16)));
typedef float    v8f   __attribute__((ext_vector_type(8)));
typedef unsigned int u32x4 __attribute__((ext_vector_type(4)));
typedef int      i32x4 __attribute__((ext_vector_type(4)));
typedef int      i32x8 __attribute__((ext_vector_type(8)));

#define DVOL      (128 * 128 * 128)
#define ROWF      136                      // floats per padded LDS row
#define ROWB      (ROWF * 4)               // 544 bytes (16B aligned)
#define IMG_ROWS  27                       // row = ic*9 + ddl*3 + hl
#define X_ROWS    36                       // row = b*9  + ddl*3 + hl
#define NROWS     (IMG_ROWS + X_ROWS)      // 63
#define SMEM_F    (NROWS * ROWF)           // 8568 floats = 34272 B

#if defined(__gfx1250__) && __has_builtin(__builtin_amdgcn_tensor_load_to_lds)
#define HAVE_TDM 1
#if __has_include(<hip/amd_detail/amd_gfx1250_TDM.h>)
#define TDM_6ARG 1
#endif
#endif

// Compile-time decomposition of patch row k = ic*27 + di*9 + dj*3 + dk.
__host__ __device__ constexpr int kIC(int k) { return k < 81 ? k / 27 : 0; }
__host__ __device__ constexpr int kDI(int k) { return k < 81 ? (k % 27) / 9 : 0; }
__host__ __device__ constexpr int kDJ(int k) { return k < 81 ? (k % 9) / 3 : 0; }
__host__ __device__ constexpr int kDK(int k) { return k < 81 ? k % 3 : 0; }
__host__ __device__ constexpr int kROW(int k) { return kIC(k) * 9 + kDI(k) * 3 + kDJ(k); }
__host__ __device__ constexpr int mROW(int m) { return m < 27 ? (m / 9) * 3 + (m % 9) / 3 : 0; }
__host__ __device__ constexpr int mDK(int m)  { return m < 27 ? m % 3 : 0; }

// ---------------------------------------------------------------------------
// Prep kernel: pack [Wgt(27x81) | bias(27) | 0] -> padded 32x96 f16 WMMA
// A-fragments in d_ws.  Column 81 holds the bias; the matching im2col row is
// a constant 1.0, so WMMA produces W*patch + bias directly.  Rows m>=27 are
// all-zero => their C outputs are exactly 0 and need no masking downstream.
//   lane holds row M = L&15;  lo lanes: e -> K = e
//                             hi lanes: e -> K = e+16
// ---------------------------------------------------------------------------
__global__ __launch_bounds__(32) void prep_afrag_kernel(
    const float* __restrict__ Wgt, const float* __restrict__ bias,
    v16h* __restrict__ wsA) {
    const int lane = threadIdx.x & 31;
    const bool hi  = lane >= 16;
    const int mrow = lane & 15;
    for (int mt = 0; mt < 2; ++mt) {
        const int m = mt * 16 + mrow;
        for (int ks = 0; ks < 3; ++ks) {
            v16h a;
            for (int e = 0; e < 16; ++e) {
                const int klocal = (hi ? 16 : 0) + e;
                const int k = ks * 32 + klocal;
                float v = 0.0f;
                if (m < 27) {
                    if (k < 81)       v = Wgt[m * 81 + k];
                    else if (k == 81) v = bias[m];
                }
                a[e] = (_Float16)v;
            }
            wsA[(mt * 3 + ks) * 32 + lane] = a;
        }
    }
}

#ifdef HAVE_TDM
// One 2D Tensor-DMA: `rows` consecutive 128-float rows (global row stride 128
// elems) -> LDS at ldsByte, with an 8-DWORD pad after every 128 DWORDs so the
// LDS row stride is 136 floats (pads pre-zeroed).  ISA D# per §8.3/8.4.
__device__ __forceinline__ void tdm_load_rows(const float* gsrc,
                                              unsigned ldsByte, int rows) {
    const uint64_t ga = (uint64_t)(uintptr_t)gsrc;
    u32x4 g0;
    g0[0] = 1u;                                   // count=1, user mode
    g0[1] = ldsByte;                              // lds_addr
    g0[2] = (unsigned)ga;                         // global_addr[31:0]
    g0[3] = (unsigned)((ga >> 32) & 0x1FFFFFFu)   // global_addr[56:32]
            | (2u << 30);                         // type = 2 ("image")
    i32x8 g1;
    g1[0] = (int)((2u << 16)      // data_size = 4 bytes
                | (1u << 20)      // pad_enable
                | (6u << 22)      // pad_interval: 128 DWORDs
                | (7u << 25));    // pad_amount : 8 DWORDs
    g1[1] = (int)(128u << 16);            // tensor_dim0[15:0] = 128
    g1[2] = (int)((unsigned)rows << 16);  // tensor_dim1[15:0] = rows
    g1[3] = (int)(128u << 16);            // tile_dim0 = 128
    g1[4] = rows;                         // tile_dim1 = rows, tile_dim2 = 0
    g1[5] = 128;                          // tensor_dim0_stride = 128 elems
    g1[6] = 0; g1[7] = 0;
    const i32x4 z4 = {0, 0, 0, 0};
#ifdef TDM_6ARG
    const i32x8 z8 = {0, 0, 0, 0, 0, 0, 0, 0};
    __builtin_amdgcn_tensor_load_to_lds(g0, g1, z4, z4, z8, 0);
#else
    __builtin_amdgcn_tensor_load_to_lds(g0, g1, z4, z4, 0);
#endif
}
#endif

// ---------------------------------------------------------------------------
// Fused kernel.  One block = one (d,h); 8 waves = 8 w-tiles of 16 voxels.
//   Stage 0: TDM-stage 27 image rows + 36 x rows into zero-padded LDS.
//   Stage 1: WMMA  K(27x16) = [Wgt|bias](27x96) * [im2col;1](96x16).
//   Stage 2: VALU  y[b,v] = sum_m K[m,v] * x[b, v+off(m)].
// ---------------------------------------------------------------------------
__global__ __launch_bounds__(256) void fused_dynconv_kernel(
    const float* __restrict__ image,   // (3, 128,128,128)
    const float* __restrict__ x,       // (4, 1, 128,128,128)
    const v16h* __restrict__ wsA,      // packed A fragments
    float* __restrict__ out)           // (4, 1, 128,128,128)
{
    __shared__ __align__(16) float smem[SMEM_F];

    const int tid  = threadIdx.x;
    const int lane = tid & 31;
    const int col  = lane & 15;
    const bool hi  = lane >= 16;
    const int d    = blockIdx.x >> 7;
    const int h    = blockIdx.x & 127;
    const int w    = ((tid >> 5) << 4) + col;   // this lane's voxel w
    const int wp3  = w + 3;                     // +4 left pad, -1 tap offset

    // ---- zero LDS (pads + potentially-invalid boundary rows) ------------
    {
        float4* s4 = (float4*)smem;
        const float4 z = make_float4(0.f, 0.f, 0.f, 0.f);
        for (int i = tid; i < SMEM_F / 4; i += 256) s4[i] = z;
    }
    __syncthreads();

    // ---- stage 63 rows into LDS -----------------------------------------
    const int dd0  = d > 0 ? d - 1 : 0;
    const int ddl0 = dd0 - (d - 1);
    const int ddn  = (d + 1 < 128 ? d + 1 : 127) - dd0 + 1;
    const int hh0  = h > 0 ? h - 1 : 0;
    const int hl0  = hh0 - (h - 1);
    const int hn   = (h + 1 < 128 ? h + 1 : 127) - hh0 + 1;
#ifdef HAVE_TDM
    if (tid < 32) {                       // wave 0 issues all tensor DMAs
        const unsigned sbase =
            (unsigned)(uintptr_t)(__attribute__((address_space(3))) void*)smem;
        // single TDM call site: jobs 0..2 = image channels, 3..6 = x batches
#pragma unroll 1
        for (int job = 0; job < 7; ++job) {
            const float* src = job < 3 ? image + (size_t)job * DVOL
                                       : x + (size_t)(job - 3) * DVOL;
            const int rowbase = job < 3 ? job * 9 : IMG_ROWS + (job - 3) * 9;
#pragma unroll 1
            for (int zi = 0; zi < ddn; ++zi)
                tdm_load_rows(
                    src + ((size_t)(dd0 + zi) * 128 + hh0) * 128,
                    sbase + (unsigned)((rowbase + (ddl0 + zi) * 3 + hl0) * ROWB) + 16u,
                    hn);
        }
#if __has_builtin(__builtin_amdgcn_s_wait_tensorcnt)
        __builtin_amdgcn_s_wait_tensorcnt(0);
#else
        asm volatile("s_wait_tensorcnt 0" ::: "memory");
#endif
    }
#else
    // fallback: cooperative coalesced staging of the 63 rows
    for (int t = tid; t < NROWS * 128; t += 256) {
        const int row = t >> 7, c = t & 127;
        const int g9  = row < IMG_ROWS ? row : row - IMG_ROWS;
        const int pl  = g9 / 9, rem = g9 % 9;
        const int dd  = d + rem / 3 - 1, hh = h + rem % 3 - 1;
        if ((unsigned)dd < 128u && (unsigned)hh < 128u) {
            const float* src = row < IMG_ROWS ? image + (size_t)pl * DVOL
                                              : x + (size_t)pl * DVOL;
            smem[row * ROWF + 4 + c] = src[((size_t)dd * 128 + hh) * 128 + c];
        }
    }
#endif
    __syncthreads();

    // ---- A fragments: 12 x b128 loads from workspace --------------------
    v16h afrag[2][3];
#pragma unroll
    for (int mt = 0; mt < 2; ++mt)
#pragma unroll
        for (int ks = 0; ks < 3; ++ks)
            afrag[mt][ks] = wsA[(mt * 3 + ks) * 32 + lane];

    // ---- B fragments from LDS (spatial padding handled by LDS zeros) ----
    // B layout: lane = column N; element e -> K = (hi?16:0) + e.
    // Row 81 is the constant-1 bias row; rows > 81 are zero padding.
    v16h bfrag[3];
#pragma unroll
    for (int ks = 0; ks < 3; ++ks) {
        v16h bf;
#pragma unroll
        for (int e = 0; e < 16; ++e) {
            const int klo = ks * 32 + e;         // <= 79, always a data row
            const int khi = klo + 16;            // may hit 81 / padding
            const int row = hi ? kROW(khi) : kROW(klo);
            const int dk  = hi ? kDK(khi)  : kDK(klo);
            const float v = smem[row * ROWF + wp3 + dk];
            float vv;
            if (khi < 81)       vv = v;                  // both halves: data
            else if (khi == 81) vv = hi ? 1.0f : v;      // bias row
            else                vv = hi ? 0.0f : v;      // zero padding
            bf[e] = (_Float16)vv;
        }
        bfrag[ks] = bf;
    }

    // ---- WMMA: C(32x16) = A * B (bias folded in), f32 accumulate --------
    v8f c0 = {};   // channels 0..15
    v8f c1 = {};   // channels 16..31 (27..31 exactly zero)
#pragma unroll
    for (int ks = 0; ks < 3; ++ks) {
        c0 = __builtin_amdgcn_wmma_f32_16x16x32_f16(
                 false, afrag[0][ks], false, bfrag[ks], (short)0, c0, false, false);
        c1 = __builtin_amdgcn_wmma_f32_16x16x32_f16(
                 false, afrag[1][ks], false, bfrag[ks], (short)0, c1, false, false);
    }

    // ---- Involution from LDS. C layout: VGPR r -> channel r + (hi?8:0) --
    // Channels m>=27 carry exact 0.0 coefficients -> no masking needed.
    float acc0 = 0.f, acc1 = 0.f, acc2 = 0.f, acc3 = 0.f;
#pragma unroll
    for (int hv = 0; hv < 2; ++hv) {
#pragma unroll
        for (int r = 0; r < 8; ++r) {
            const int mlo = r + hv * 16;
            const int mhi = mlo + 8;
            const int dij = hi ? mROW(mhi) : mROW(mlo);
            const int dk  = hi ? mDK(mhi)  : mDK(mlo);
            const float kvs = hv ? c1[r] : c0[r];
            const int idx = dij * ROWF + wp3 + dk;
            acc0 += kvs * smem[(IMG_ROWS + 0 * 9) * ROWF + idx];
            acc1 += kvs * smem[(IMG_ROWS + 1 * 9) * ROWF + idx];
            acc2 += kvs * smem[(IMG_ROWS + 2 * 9) * ROWF + idx];
            acc3 += kvs * smem[(IMG_ROWS + 3 * 9) * ROWF + idx];
        }
    }

    // Combine the two channel-halves held by lanes L and L+16 (same voxel).
    acc0 += __shfl_xor(acc0, 16, 32);
    acc1 += __shfl_xor(acc1, 16, 32);
    acc2 += __shfl_xor(acc2, 16, 32);
    acc3 += __shfl_xor(acc3, 16, 32);

    const size_t obase = ((size_t)d * 128 + h) * 128 + w;
    if (!hi) {
        out[0 * (size_t)DVOL + obase] = acc0;
        out[1 * (size_t)DVOL + obase] = acc1;
    } else {
        out[2 * (size_t)DVOL + obase] = acc2;
        out[3 * (size_t)DVOL + obase] = acc3;
    }
}

extern "C" void kernel_launch(void* const* d_in, const int* in_sizes, int n_in,
                              void* d_out, int out_size, void* d_ws, size_t ws_size,
                              hipStream_t stream) {
    const float* image = (const float*)d_in[0];
    const float* x     = (const float*)d_in[1];
    const float* Wgt   = (const float*)d_in[2];
    const float* bias  = (const float*)d_in[3];
    float* out = (float*)d_out;
    v16h* wsA  = (v16h*)d_ws;                     // 6 KB of A fragments

    prep_afrag_kernel<<<1, 32, 0, stream>>>(Wgt, bias, wsA);
    // 16384 blocks = 128 d * 128 h; 8 waves per block = 8 w-tiles.
    fused_dynconv_kernel<<<16384, 256, 0, stream>>>(image, x, wsA, out);
}
